// GATLayer_22351009808408
// MI455X (gfx1250) — matrix operand order repaired and checked by reading, not compiled, not run on hardware
//
#include <hip/hip_runtime.h>

#define C_DIM 128
#define LDA   136          // padded LDS row stride (ushorts): 272B = 68 dwords -> conflict-free
#define NEG_SLOPE 0.2f

typedef __attribute__((ext_vector_type(16))) __bf16 v16bf;
typedef __attribute__((ext_vector_type(8)))  float  v8f;

union BF16Frag { uint4 q[2]; unsigned short u[16]; v16bf v; };

__device__ __forceinline__ unsigned short f2bf(float f) {
    union { float f; unsigned u; } c; c.f = f;
    unsigned r = c.u + 0x7FFFu + ((c.u >> 16) & 1u);   // round-to-nearest-even
    return (unsigned short)(r >> 16);
}

// ---------------------------------------------------------------------------
// One-time: WT[n*LDA + k] = bf16(W[k,n])  (padded, K-contiguous per column)
// ---------------------------------------------------------------------------
__global__ void k_prep_wt(const float* __restrict__ W, unsigned short* __restrict__ WT)
{
    int i = blockIdx.x * blockDim.x + threadIdx.x;   // i = k*128 + n
    if (i < C_DIM * C_DIM) {
        int k = i >> 7, n = i & 127;
        WT[n * LDA + k] = f2bf(W[i]);
    }
}

// ---------------------------------------------------------------------------
// GEMM: H[n,128] = X[n,128] @ W[128,128]   (bf16 WMMA, f32 accumulate)
// Block = 256 threads = 8 waves. Block tile = 128 rows x 128 cols.
// Wave w owns rows [w*16, w*16+16), all 128 cols -> 8 v8f accumulators.
// All fragments load as 2x ds_load_b128 from padded LDS.
// ---------------------------------------------------------------------------
__global__ __launch_bounds__(256) void k_gemm(const float* __restrict__ X,
                                              const unsigned short* __restrict__ WT,
                                              float* __restrict__ H, int nrows)
{
    __shared__ __align__(16) unsigned short sA[128 * LDA];    // x tile [m][k] bf16
    __shared__ __align__(16) unsigned short sBT[C_DIM * LDA]; // W^T    [n][k] bf16

    const int tid     = threadIdx.x;
    const int wave    = tid >> 5;
    const int lane    = tid & 31;
    const int rowBase = blockIdx.x * 128;

    // stage X tile: float4 load -> 4x bf16 -> packed uint2 (ds_store_b64)
    for (int q = tid; q < 128 * 32; q += 256) {       // 32 float4 per row
        int m = q >> 5, k4 = (q & 31) * 4;
        int gr = rowBase + m;
        float4 f = (gr < nrows) ? ((const float4*)(X + (size_t)gr * C_DIM))[q & 31]
                                : make_float4(0.f, 0.f, 0.f, 0.f);
        uint2 p;
        p.x = (unsigned)f2bf(f.x) | ((unsigned)f2bf(f.y) << 16);
        p.y = (unsigned)f2bf(f.z) | ((unsigned)f2bf(f.w) << 16);
        *(uint2*)(sA + m * LDA + k4) = p;
    }
    // stage W^T: straight linear uint4 copy (already converted + padded)
    {
        const uint4* src = (const uint4*)WT;
        uint4* dst = (uint4*)sBT;
        const int nq = C_DIM * LDA * 2 / 16;          // 2176 uint4
        for (int i = tid; i < nq; i += 256) dst[i] = src[i];
    }
    __syncthreads();

    const int m0    = wave * 16;        // wave's row tile inside block
    const int mrow  = lane & 15;        // A: M = lane % 16
    const int khalf = (lane >> 4) * 8;  // A: K-half selector per ISA layout

    v8f acc[8] = {};

    for (int kb = 0; kb < 4; ++kb) {
        const int kbase = kb * 32;

        // A fragment 16x32 bf16 (ISA 7.12.2): two contiguous 16B runs
        BF16Frag afr;
        const uint4* ap = (const uint4*)(sA + (m0 + mrow) * LDA + kbase + khalf);
        afr.q[0] = ap[0];       // K = kbase+khalf   .. +7
        afr.q[1] = ap[2];       // K = kbase+16+khalf.. +7 (skip 16 ushorts)

        // B fragments 32x16 bf16: lane%16 = N, same K layout as A (khalf, khalf+16);
        // K-contiguous in sBT -> two 16B runs per lane
        const int kb0 = kbase + (lane >> 4) * 8;
        #pragma unroll
        for (int t = 0; t < 8; ++t) {
            BF16Frag bfr;
            const uint4* bp = (const uint4*)(sBT + (t * 16 + (lane & 15)) * LDA + kb0);
            bfr.q[0] = bp[0];
            bfr.q[1] = bp[2];
            acc[t] = __builtin_amdgcn_wmma_f32_16x16x32_bf16(
                false, afr.v, false, bfr.v, (short)0, acc[t], false, false);
        }
    }

    // C/D layout: VGPR j -> M = j + 8*(lane>=16), N = lane%16
    const int rtop = (lane >> 4) * 8;
    #pragma unroll
    for (int t = 0; t < 8; ++t) {
        #pragma unroll
        for (int j = 0; j < 8; ++j) {
            int gr = rowBase + m0 + rtop + j;
            if (gr < nrows)
                H[(size_t)gr * C_DIM + t * 16 + (lane & 15)] = acc[t][j];
        }
    }
}

// ---------------------------------------------------------------------------
// alpha_s[i] = h[i,:] . a_src ; alpha_d[i] = h[i,:] . a_dst  (one wave / node)
// ---------------------------------------------------------------------------
__global__ __launch_bounds__(256) void k_alpha(const float* __restrict__ H,
                                               const float* __restrict__ av_s,
                                               const float* __restrict__ av_d,
                                               float* __restrict__ As,
                                               float* __restrict__ Ad, int n)
{
    int row  = (blockIdx.x * 256 + threadIdx.x) >> 5;
    int lane = threadIdx.x & 31;
    if (row >= n) return;
    float4 hv = ((const float4*)(H + (size_t)row * C_DIM))[lane];
    float4 s4 = ((const float4*)av_s)[lane];
    float4 d4 = ((const float4*)av_d)[lane];
    float ss = hv.x * s4.x + hv.y * s4.y + hv.z * s4.z + hv.w * s4.w;
    float dd = hv.x * d4.x + hv.y * d4.y + hv.z * d4.z + hv.w * d4.w;
    #pragma unroll
    for (int off = 16; off; off >>= 1) {
        ss += __shfl_xor(ss, off, 32);
        dd += __shfl_xor(dd, off, 32);
    }
    if (lane == 0) { As[row] = ss; Ad[row] = dd; }
}

// ---------------------------------------------------------------------------
// per-segment init + bias fill
// ---------------------------------------------------------------------------
__global__ void k_fill_seg(float* m, float* den, int n) {
    int i = blockIdx.x * blockDim.x + threadIdx.x;
    if (i < n) { m[i] = -__builtin_inff(); den[i] = 0.0f; }
}
__global__ void k_fill_bias(float* out, const float* __restrict__ b, int total) {
    int i = blockIdx.x * blockDim.x + threadIdx.x;
    if (i < total) out[i] = b[i & (C_DIM - 1)];
}

// sign-split trick: float max via int max / uint min (correct across mixed signs)
__device__ __forceinline__ void atomicMaxF(float* addr, float v) {
    if (v >= 0.0f) atomicMax((int*)addr, __float_as_int(v));
    else           atomicMin((unsigned int*)addr, __float_as_uint(v));
}

// ---------------------------------------------------------------------------
// edge logits + leaky relu + segment max
// ---------------------------------------------------------------------------
__global__ void k_logit(const float* __restrict__ As, const float* __restrict__ Ad,
                        const int* __restrict__ src, const int* __restrict__ dst,
                        float* __restrict__ elog, float* __restrict__ m,
                        int E_, int EP)
{
    int e = blockIdx.x * blockDim.x + threadIdx.x;
    if (e >= EP) return;
    int s = (e < E_) ? src[e] : (e - E_);   // self loops appended
    int d = (e < E_) ? dst[e] : (e - E_);
    float l = As[s] + Ad[d];
    l = (l > 0.0f) ? l : NEG_SLOPE * l;
    elog[e] = l;
    atomicMaxF(&m[d], l);
}

// ---------------------------------------------------------------------------
// ex = exp(logit - m[dst]); denom[dst] += ex
// ---------------------------------------------------------------------------
__global__ void k_expsum(const float* __restrict__ elog, const float* __restrict__ m,
                         const int* __restrict__ dst, float* __restrict__ ex,
                         float* __restrict__ den, int E_, int EP)
{
    int e = blockIdx.x * blockDim.x + threadIdx.x;
    if (e >= EP) return;
    int d = (e < E_) ? dst[e] : (e - E_);
    float v = expf(elog[e] - m[d]);
    ex[e] = v;
    atomicAdd(&den[d], v);
}

// ---------------------------------------------------------------------------
// out[dst,:] += (ex/denom) * h[src,:]   (one wave per edge, 4 ch per lane)
// ---------------------------------------------------------------------------
__global__ __launch_bounds__(256) void k_aggregate(
    const float* __restrict__ ex, const float* __restrict__ den,
    const float* __restrict__ H, const int* __restrict__ src,
    const int* __restrict__ dst, float* __restrict__ out, int E_, int EP)
{
    int e    = (blockIdx.x * 256 + threadIdx.x) >> 5;
    int lane = threadIdx.x & 31;
    if (e >= EP) return;
    int s = (e < E_) ? src[e] : (e - E_);
    int d = (e < E_) ? dst[e] : (e - E_);
    float coef = ex[e] / fmaxf(den[d], 1e-16f);
    float4 hv = ((const float4*)(H + (size_t)s * C_DIM))[lane];
    float* op = out + (size_t)d * C_DIM + lane * 4;
    atomicAdd(op + 0, coef * hv.x);
    atomicAdd(op + 1, coef * hv.y);
    atomicAdd(op + 2, coef * hv.z);
    atomicAdd(op + 3, coef * hv.w);
}

__global__ void k_relu(float* out, int total) {
    int i = blockIdx.x * blockDim.x + threadIdx.x;
    if (i < total) out[i] = fmaxf(out[i], 0.0f);
}

// ---------------------------------------------------------------------------
// host side
// ---------------------------------------------------------------------------
static void run_layer(const float* X, const int* src, const int* dst,
                      const float* W, const float* avs, const float* avd,
                      const float* b, unsigned short* WT, float* H, float* OUT,
                      float* As, float* Ad, float* M, float* DEN,
                      float* ELOG, float* EX, int n, int E_, int EP,
                      hipStream_t stream)
{
    k_prep_wt  <<<(C_DIM * C_DIM + 255) / 256, 256, 0, stream>>>(W, WT);
    k_fill_seg <<<(n + 255) / 256, 256, 0, stream>>>(M, DEN, n);
    k_fill_bias<<<(n * C_DIM + 255) / 256, 256, 0, stream>>>(OUT, b, n * C_DIM);
    k_gemm     <<<(n + 127) / 128, 256, 0, stream>>>(X, WT, H, n);
    k_alpha    <<<(n + 7) / 8, 256, 0, stream>>>(H, avs, avd, As, Ad, n);
    k_logit    <<<(EP + 255) / 256, 256, 0, stream>>>(As, Ad, src, dst, ELOG, M, E_, EP);
    k_expsum   <<<(EP + 255) / 256, 256, 0, stream>>>(ELOG, M, dst, EX, DEN, E_, EP);
    k_aggregate<<<(EP + 7) / 8, 256, 0, stream>>>(EX, DEN, H, src, dst, OUT, E_, EP);
}

extern "C" void kernel_launch(void* const* d_in, const int* in_sizes, int n_in,
                              void* d_out, int out_size, void* d_ws, size_t ws_size,
                              hipStream_t stream)
{
    const float* x   = (const float*)d_in[0];
    const int*   e   = (const int*)  d_in[1];
    const float* W1  = (const float*)d_in[2];
    const float* as1 = (const float*)d_in[3];
    const float* ad1 = (const float*)d_in[4];
    const float* b1  = (const float*)d_in[5];
    const float* W2  = (const float*)d_in[6];
    const float* as2 = (const float*)d_in[7];
    const float* ad2 = (const float*)d_in[8];
    const float* b2  = (const float*)d_in[9];
    float* out = (float*)d_out;

    const int n  = in_sizes[0] / C_DIM;     // 100000
    const int E_ = in_sizes[1] / 2;         // 1600000
    const int EP = E_ + n;                  // with self loops
    const int* src = e;
    const int* dst = e + E_;

    auto al = [](size_t x_) { return (x_ + 255) & ~(size_t)255; };
    char* w = (char*)d_ws;
    float* H    = (float*)w; w += al((size_t)n * C_DIM * 4);
    float* OUT1 = (float*)w; w += al((size_t)n * C_DIM * 4);
    float* As   = (float*)w; w += al((size_t)n * 4);
    float* Ad   = (float*)w; w += al((size_t)n * 4);
    float* M    = (float*)w; w += al((size_t)n * 4);
    float* DEN  = (float*)w; w += al((size_t)n * 4);
    float* ELOG = (float*)w; w += al((size_t)EP * 4);
    float* EX   = (float*)w; w += al((size_t)EP * 4);
    unsigned short* WT = (unsigned short*)w; w += al((size_t)C_DIM * LDA * 2);

    // layer 1: x -> OUT1
    run_layer(x, src, dst, W1, as1, ad1, b1, WT, H, OUT1,
              As, Ad, M, DEN, ELOG, EX, n, E_, EP, stream);
    // layer 2: OUT1 -> d_out
    run_layer(OUT1, src, dst, W2, as2, ad2, b2, WT, H, out,
              As, Ad, M, DEN, ELOG, EX, n, E_, EP, stream);
    // final relu
    k_relu<<<((size_t)n * C_DIM + 255) / 256, 256, 0, stream>>>(out, n * C_DIM);
}
